// HAN_1503238553910
// MI455X (gfx1250) — matrix operand order repaired and checked by reading, not compiled, not run on hardware
//
#include <hip/hip_runtime.h>
#include <hip/hip_bf16.h>
#include <math.h>

#define N_NODES   50000
#define N_EDGES   400000
#define DIM       128
#define HID       32
#define NHEADS    4
#define OUT_DIM   32
#define NTYPES    3
#define LDZ       (NTYPES * DIM)   // 384, row stride of z[N][T][128]

typedef _Float16 half_t;
typedef __attribute__((ext_vector_type(16))) half_t v16h;
typedef __attribute__((ext_vector_type(8)))  half_t v8h;
typedef __attribute__((ext_vector_type(8)))  float  v8f;

// ---------------------------------------------------------------------------
// Ordered-int encoding so signed-int atomicMax == float max
// ---------------------------------------------------------------------------
__device__ __forceinline__ int f2oi(float f) {
    int i = __float_as_int(f);
    return (i >= 0) ? i : (i ^ 0x7FFFFFFF);
}
__device__ __forceinline__ float oi2f(int i) {
    return __int_as_float((i >= 0) ? i : (i ^ 0x7FFFFFFF));
}

// ---------------------------------------------------------------------------
// WMMA GEMM with async-copy double buffering:
//   C[M,Ncols] = act(A[M,K]_f16 @ Bt[Ncols,K]_f16^T + bias)
// A packed f16 (row stride lda elems), Bt pre-transposed f16 [Ncols][K].
// Block: 256 threads = 8 waves; tile 128(M) x 64(N); K stepped by 32.
// HBM->LDS staging uses global_load_async_to_lds_b128 (ASYNCcnt), double
// buffered so tile k+1 streams in while tile k feeds v_wmma.
// act: 0 = identity, 1 = tanh
// ---------------------------------------------------------------------------
#define TM 128
#define TN 64
#define TK 32
#define LDS_STRIDE 40   // halfs per row: 80 B, multiple of 16 B

__global__ __launch_bounds__(256) void gemm_wmma_async(
    const half_t* __restrict__ A, int lda,
    const half_t* __restrict__ Bt,            // [Ncols][K]
    const float* __restrict__ bias,
    float* __restrict__ C, int ldc,
    int M, int Ncols, int K, int act)
{
    __shared__ __align__(16) half_t As[2][TM][LDS_STRIDE];
    __shared__ __align__(16) half_t Bs[2][TN][LDS_STRIDE];

    const int tid  = threadIdx.x;
    const int lane = tid & 31;
    const int wave = tid >> 5;                 // 0..7
    const int m0 = blockIdx.x * TM;
    const int n0 = blockIdx.y * TN;

    const half_t* Abase = A  + (long)m0 * lda; // uniform
    const half_t* Bbase = Bt + (long)n0 * K;   // uniform

    v8f acc[4];
    #pragma unroll
    for (int i = 0; i < 4; ++i)
        #pragma unroll
        for (int j = 0; j < 8; ++j) acc[i][j] = 0.0f;

    // issue async copy of the K-tile starting at k0 into LDS buffer b
    auto stage = [&](int k0, int b) {
        // A tile: 128 rows x 64 B  = 512 x 16B chunks, 2 per thread
        #pragma unroll
        for (int c = 0; c < 2; ++c) {
            const int chunk = tid * 2 + c;          // 0..511
            const int row = chunk >> 2;             // 0..127
            const int col = chunk & 3;              // 16B columns
            unsigned lds  = (unsigned)(uintptr_t)&As[b][row][col * 8];
            unsigned goff = (unsigned)((row * lda + k0 + col * 8) * (int)sizeof(half_t));
            asm volatile("global_load_async_to_lds_b128 %0, %1, %2"
                         :: "v"(lds), "v"(goff), "s"(Abase) : "memory");
        }
        // B tile: 64 rows x 64 B = 256 x 16B chunks, 1 per thread
        {
            const int row = tid >> 2;               // 0..63
            const int col = tid & 3;
            unsigned lds  = (unsigned)(uintptr_t)&Bs[b][row][col * 8];
            unsigned goff = (unsigned)((row * K + k0 + col * 8) * (int)sizeof(half_t));
            asm volatile("global_load_async_to_lds_b128 %0, %1, %2"
                         :: "v"(lds), "v"(goff), "s"(Bbase) : "memory");
        }
    };

    stage(0, 0);
    asm volatile("s_wait_asynccnt 0x0" ::: "memory");
    __syncthreads();

    const int nk = K / TK;
    for (int kt = 0; kt < nk; ++kt) {
        const int cur = kt & 1;
        if (kt + 1 < nk) stage((kt + 1) * TK, cur ^ 1);

        // ---- A fragment (16x32 f16, CDNA5 lane layout), two b128 LDS loads
        v16h afrag;
        {
            const int r  = wave * 16 + (lane & 15);
            const int kh = (lane >> 4) * 8;
            const half_t* ap = &As[cur][r][0];
            v8h lo = *(const v8h*)(ap + kh);
            v8h hi = *(const v8h*)(ap + kh + 16);
            afrag = __builtin_shufflevector(lo, hi, 0, 1, 2, 3, 4, 5, 6, 7,
                                            8, 9, 10, 11, 12, 13, 14, 15);
        }
        // ---- 4 N-tiles: B fragment (32x16) + WMMA
        #pragma unroll
        for (int nt = 0; nt < 4; ++nt) {
            const int n  = nt * 16 + (lane & 15);
            const int kb = (lane >> 4) * 8;
            const half_t* bp = &Bs[cur][n][0];
            v8h b0 = *(const v8h*)(bp + kb);
            v8h b1 = *(const v8h*)(bp + kb + 16);
            v16h bfrag = __builtin_shufflevector(b0, b1, 0, 1, 2, 3, 4, 5, 6, 7,
                                                 8, 9, 10, 11, 12, 13, 14, 15);
            acc[nt] = __builtin_amdgcn_wmma_f32_16x16x32_f16(
                false, afrag, false, bfrag, (short)0, acc[nt], false, false);
        }

        asm volatile("s_wait_asynccnt 0x0" ::: "memory");
        __syncthreads();
    }

    // ---- epilogue: bias + optional tanh, guarded store
    const int mbase = m0 + wave * 16 + (lane >> 4) * 8;
    const int nlane = lane & 15;
    #pragma unroll
    for (int nt = 0; nt < 4; ++nt) {
        const int n = n0 + nt * 16 + nlane;
        if (n >= Ncols) continue;
        const float bv = bias[n];
        #pragma unroll
        for (int r = 0; r < 8; ++r) {
            const int m = mbase + r;
            if (m < M) {
                float v = acc[nt][r] + bv;
                if (act == 1) v = tanhf(v);
                C[(long)m * ldc + n] = v;
            }
        }
    }
}

// ---------------------------------------------------------------------------
// Utility fills / conversions
// ---------------------------------------------------------------------------
__global__ void set_f32_kernel(float* p, float v, long n) {
    long i = (long)blockIdx.x * blockDim.x + threadIdx.x;
    if (i < n) p[i] = v;
}
__global__ void set_i32_kernel(int* p, int v, long n) {
    long i = (long)blockIdx.x * blockDim.x + threadIdx.x;
    if (i < n) p[i] = v;
}
__global__ void cvt_f16_kernel(const float* __restrict__ src, half_t* __restrict__ dst, long n) {
    long i = (long)blockIdx.x * blockDim.x + threadIdx.x;
    if (i < n) dst[i] = (half_t)src[i];
}
// Wt[n][k] = (f16) W[k][n]   (W is [K][Ncols] row-major)
__global__ void wt_f16_kernel(const float* __restrict__ W, half_t* __restrict__ Wt,
                              int K, int Ncols) {
    const int idx = blockIdx.x * blockDim.x + threadIdx.x;
    if (idx >= K * Ncols) return;
    const int n = idx / K, k = idx - n * K;
    Wt[idx] = (half_t)W[(long)k * Ncols + n];
}

// ---------------------------------------------------------------------------
// Edge pass 1: logit[e,h] = sum_d attn[h,d]*leaky(fs[src]+fd[dst]); seg-max
// ---------------------------------------------------------------------------
__global__ __launch_bounds__(256) void edge_logits_kernel(
    const int* __restrict__ src, const int* __restrict__ dst,
    const float* __restrict__ fs, const float* __restrict__ fd,
    const float* __restrict__ attn, float* __restrict__ lbuf,
    int* __restrict__ mbuf)
{
    const int idx = blockIdx.x * blockDim.x + threadIdx.x;
    if (idx >= N_EDGES * NHEADS) return;
    const int e = idx >> 2, h = idx & 3;
    const int s = src[e], d = dst[e];
    const float4* ps = (const float4*)(fs + (long)s * DIM + h * HID);
    const float4* pd = (const float4*)(fd + (long)d * DIM + h * HID);
    const float4* pa = (const float4*)(attn + h * HID);
    float acc = 0.0f;
    #pragma unroll
    for (int q = 0; q < 8; ++q) {
        float4 a = ps[q], b = pd[q], w = pa[q];
        float v;
        v = a.x + b.x; v = v > 0.0f ? v : 0.2f * v; acc += v * w.x;
        v = a.y + b.y; v = v > 0.0f ? v : 0.2f * v; acc += v * w.y;
        v = a.z + b.z; v = v > 0.0f ? v : 0.2f * v; acc += v * w.z;
        v = a.w + b.w; v = v > 0.0f ? v : 0.2f * v; acc += v * w.w;
    }
    lbuf[idx] = acc;
    atomicMax(mbuf + (long)d * NHEADS + h, f2oi(acc));
}

// m = decode(mbuf) (0 for empty segments); den = 0
__global__ void finalize_max_kernel(const int* __restrict__ mbuf,
                                    float* __restrict__ m, float* __restrict__ den, long n)
{
    long i = (long)blockIdx.x * blockDim.x + threadIdx.x;
    if (i >= n) return;
    const int v = mbuf[i];
    m[i]   = (v == (int)0x80000000) ? 0.0f : oi2f(v);
    den[i] = 0.0f;
}

// Edge pass 2: ex = exp(logit - m[dst]); seg-sum into den
__global__ __launch_bounds__(256) void edge_exp_kernel(
    const int* __restrict__ dst, const float* __restrict__ m,
    float* __restrict__ lbuf, float* __restrict__ den)
{
    const int idx = blockIdx.x * blockDim.x + threadIdx.x;
    if (idx >= N_EDGES * NHEADS) return;
    const int e = idx >> 2, h = idx & 3;
    const int d = dst[e];
    const float ex = __expf(lbuf[idx] - m[(long)d * NHEADS + h]);
    lbuf[idx] = ex;
    atomicAdd(den + (long)d * NHEADS + h, ex);
}

// Edge pass 3: z[dst] += (ex/den[dst]) * fs[src]
__global__ __launch_bounds__(256) void edge_aggregate_kernel(
    const int* __restrict__ src, const int* __restrict__ dst,
    const float* __restrict__ fs, const float* __restrict__ lbuf,
    const float* __restrict__ den, float* __restrict__ zt)
{
    const int idx = blockIdx.x * blockDim.x + threadIdx.x;
    if (idx >= N_EDGES * NHEADS) return;
    const int e = idx >> 2, h = idx & 3;
    const int s = src[e], d = dst[e];
    const float a = lbuf[idx] / fmaxf(den[(long)d * NHEADS + h], 1e-16f);
    const float4* ps = (const float4*)(fs + (long)s * DIM + h * HID);
    float* po = zt + (long)d * LDZ + h * HID;
    #pragma unroll
    for (int q = 0; q < 8; ++q) {
        float4 v = ps[q];
        atomicAdd(po + q * 4 + 0, a * v.x);
        atomicAdd(po + q * 4 + 1, a * v.y);
        atomicAdd(po + q * 4 + 2, a * v.z);
        atomicAdd(po + q * 4 + 3, a * v.w);
    }
}

// ELU in place on one type slice of z (row stride LDZ); also emit f16 copy
__global__ void elu_kernel(float* __restrict__ zt, half_t* __restrict__ zht) {
    const long idx = (long)blockIdx.x * blockDim.x + threadIdx.x;
    if (idx >= (long)N_NODES * DIM) return;
    const long n = idx >> 7;        // / 128
    const int  c = (int)(idx & 127);
    float* p = zt + n * LDZ + c;
    float v = *p;
    v = (v > 0.0f) ? v : (__expf(v) - 1.0f);
    *p = v;
    zht[n * LDZ + c] = (half_t)v;
}

// Per-node dot with W2, block-reduced, one atomicAdd per block -> wsum[t]
__global__ __launch_bounds__(256) void sem_dot_kernel(
    const float* __restrict__ s1, const float* __restrict__ W2,
    float* __restrict__ wsum_t)
{
    __shared__ float red[256];
    const int node = blockIdx.x * 256 + threadIdx.x;
    float w = 0.0f;
    if (node < N_NODES) {
        const float4* ps = (const float4*)(s1 + (long)node * DIM);
        const float4* pw = (const float4*)W2;
        #pragma unroll
        for (int q = 0; q < 32; ++q) {
            float4 a = ps[q], b = pw[q];
            w += a.x * b.x + a.y * b.y + a.z * b.z + a.w * b.w;
        }
    }
    red[threadIdx.x] = w;
    __syncthreads();
    for (int s = 128; s > 0; s >>= 1) {
        if (threadIdx.x < s) red[threadIdx.x] += red[threadIdx.x + s];
        __syncthreads();
    }
    if (threadIdx.x == 0) atomicAdd(wsum_t, red[0]);
}

// beta = softmax(wsum / N) over 3 types
__global__ void beta_kernel(const float* __restrict__ wsum, float* __restrict__ beta) {
    if (threadIdx.x == 0 && blockIdx.x == 0) {
        const float inv = 1.0f / (float)N_NODES;
        float a = wsum[0] * inv, b = wsum[1] * inv, c = wsum[2] * inv;
        float mx = fmaxf(a, fmaxf(b, c));
        float ea = __expf(a - mx), eb = __expf(b - mx), ec = __expf(c - mx);
        float s = ea + eb + ec;
        beta[0] = ea / s; beta[1] = eb / s; beta[2] = ec / s;
    }
}

// x16[n,d] = (f16) sum_t beta[t] * z[n,t,d]   (next layer's GEMM operand)
__global__ void combine_kernel(const float* __restrict__ z, const float* __restrict__ beta,
                               half_t* __restrict__ xh)
{
    const long idx = (long)blockIdx.x * blockDim.x + threadIdx.x;
    if (idx >= (long)N_NODES * DIM) return;
    const long n = idx >> 7;
    const int  c = (int)(idx & 127);
    const float* zr = z + n * LDZ + c;
    xh[idx] = (half_t)(beta[0] * zr[0] + beta[1] * zr[DIM] + beta[2] * zr[2 * DIM]);
}

// ---------------------------------------------------------------------------
extern "C" void kernel_launch(void* const* d_in, const int* in_sizes, int n_in,
                              void* d_out, int out_size, void* d_ws, size_t ws_size,
                              hipStream_t stream) {
    (void)in_sizes; (void)n_in; (void)out_size; (void)ws_size;

    // ---- inputs (setup_inputs dict order, pytree-flattened)
    const float* h = (const float*)d_in[0];
    const int* edge_ptr[NTYPES] = {
        (const int*)d_in[1], (const int*)d_in[2], (const int*)d_in[3] };
    // params: per layer li, per type t at base 4 + li*18 + t*5:
    //   +0 Ws, +1 bs, +2 Wd, +3 bd, +4 attn; sem: +15 W1, +16 b1, +17 W2
    // then 40 = W_out, 41 = b_out
    auto P = [&](int i) { return (const float*)d_in[i]; };

    // ---- workspace layout
    // f32 / int buffers first, async-read f16 buffers afterwards so that any
    // ragged-tile async over-read stays inside the workspace.
    float* z    = (float*)d_ws;                            // N * 384
    float* fs   = z    + (long)N_NODES * LDZ;              // N * 128 (reused as s1)
    float* fd   = fs   + (long)N_NODES * DIM;              // N * 128
    float* lbuf = fd   + (long)N_NODES * DIM;              // E * 4
    float* m    = lbuf + (long)N_EDGES * NHEADS;           // N * 4
    float* den  = m    + (long)N_NODES * NHEADS;           // N * 4
    int*   mbuf = (int*)(den + (long)N_NODES * NHEADS);    // N * 4
    float* wsum = (float*)(mbuf + (long)N_NODES * NHEADS); // 4
    float* beta = wsum + 4;                                // 4
    half_t* xh  = (half_t*)(beta + 4);                     // N * 128 (f16)
    half_t* zh  = xh + (long)N_NODES * DIM;                // N * 384 (f16)
    half_t* wt  = zh + (long)N_NODES * LDZ;                // 15 * 128*128 (f16)
    // + 64 KB tail pad after wt (async over-read safety)

    // transposed f16 weight slots: per layer li: Ws(t)=li*7+2t, Wd(t)=li*7+2t+1,
    // W1=li*7+6; W_out = slot 14
    auto WT = [&](int slot) { return wt + (long)slot * (DIM * DIM); };

    const dim3 gemm_block(256);
    const dim3 gemm_grid128((N_NODES + TM - 1) / TM, DIM / TN);      // (391, 2)
    const dim3 gemm_grid32((N_NODES + TM - 1) / TM, 1);              // (391, 1)
    const int  tpb = 256;
    const unsigned eh_blocks   = (N_EDGES * NHEADS + tpb - 1) / tpb;
    const unsigned nh_blocks   = (N_NODES * NHEADS + tpb - 1) / tpb;
    const unsigned nd_blocks   = ((long)N_NODES * DIM + tpb - 1) / tpb;
    const unsigned node_blocks = (N_NODES + tpb - 1) / tpb;
    const unsigned z_blocks    = ((long)N_NODES * LDZ + tpb - 1) / tpb;
    const unsigned w_blocks    = (DIM * DIM + tpb - 1) / tpb;
    const unsigned wo_blocks   = (DIM * OUT_DIM + tpb - 1) / tpb;

    // ---- one-time weight transposes (f32 -> f16, [Ncols][K])
    for (int li = 0; li < 2; ++li) {
        const int base = 4 + li * 18;
        for (int t = 0; t < NTYPES; ++t) {
            wt_f16_kernel<<<w_blocks, tpb, 0, stream>>>(P(base + t * 5 + 0), WT(li * 7 + 2 * t + 0), DIM, DIM);
            wt_f16_kernel<<<w_blocks, tpb, 0, stream>>>(P(base + t * 5 + 2), WT(li * 7 + 2 * t + 1), DIM, DIM);
        }
        wt_f16_kernel<<<w_blocks, tpb, 0, stream>>>(P(base + 15), WT(li * 7 + 6), DIM, DIM);
    }
    wt_f16_kernel<<<wo_blocks, tpb, 0, stream>>>(P(40), WT(14), DIM, OUT_DIM);

    // ---- layer 0 input: xh = f16(h)
    cvt_f16_kernel<<<nd_blocks, tpb, 0, stream>>>(h, xh, (long)N_NODES * DIM);

    for (int li = 0; li < 2; ++li) {
        const int base = 4 + li * 18;

        set_f32_kernel<<<z_blocks, tpb, 0, stream>>>(z, 0.0f, (long)N_NODES * LDZ);

        for (int t = 0; t < NTYPES; ++t) {
            const float* bs   = P(base + t * 5 + 1);
            const float* bd   = P(base + t * 5 + 3);
            const float* attn = P(base + t * 5 + 4);
            const int* src = edge_ptr[t];
            const int* dst = edge_ptr[t] + N_EDGES;

            // projections: fs = x@Ws+bs, fd = x@Wd+bd   (WMMA + async staging)
            gemm_wmma_async<<<gemm_grid128, gemm_block, 0, stream>>>(
                xh, DIM, WT(li * 7 + 2 * t + 0), bs, fs, DIM, N_NODES, DIM, DIM, 0);
            gemm_wmma_async<<<gemm_grid128, gemm_block, 0, stream>>>(
                xh, DIM, WT(li * 7 + 2 * t + 1), bd, fd, DIM, N_NODES, DIM, DIM, 0);

            // edge softmax + aggregation
            set_i32_kernel<<<nh_blocks, tpb, 0, stream>>>(
                mbuf, (int)0x80000000, (long)N_NODES * NHEADS);
            edge_logits_kernel<<<eh_blocks, tpb, 0, stream>>>(
                src, dst, fs, fd, attn, lbuf, mbuf);
            finalize_max_kernel<<<nh_blocks, tpb, 0, stream>>>(
                mbuf, m, den, (long)N_NODES * NHEADS);
            edge_exp_kernel<<<eh_blocks, tpb, 0, stream>>>(dst, m, lbuf, den);
            edge_aggregate_kernel<<<eh_blocks, tpb, 0, stream>>>(
                src, dst, fs, lbuf, den, z + t * DIM);
            elu_kernel<<<nd_blocks, tpb, 0, stream>>>(z + t * DIM, zh + t * DIM);
        }

        // semantic attention
        const float* b1 = P(base + 16);
        const float* W2 = P(base + 17);
        set_f32_kernel<<<1, 32, 0, stream>>>(wsum, 0.0f, 4);
        for (int t = 0; t < NTYPES; ++t) {
            // s1 = tanh(z_t @ W1 + b1)  (reuse fs buffer; WMMA, fused tanh)
            gemm_wmma_async<<<gemm_grid128, gemm_block, 0, stream>>>(
                zh + t * DIM, LDZ, WT(li * 7 + 6), b1, fs, DIM, N_NODES, DIM, DIM, 1);
            sem_dot_kernel<<<node_blocks, tpb, 0, stream>>>(fs, W2, wsum + t);
        }
        beta_kernel<<<1, 32, 0, stream>>>(wsum, beta);
        combine_kernel<<<nd_blocks, tpb, 0, stream>>>(z, beta, xh);
    }

    // final: out = x @ W_out + b_out   (WMMA, Ncols = 32)
    gemm_wmma_async<<<gemm_grid32, gemm_block, 0, stream>>>(
        xh, DIM, WT(14), P(41), (float*)d_out, OUT_DIM, N_NODES, OUT_DIM, DIM, 0);
}
